// MambaModel_47983374630969
// MI455X (gfx1250) — compile-verified, matrix operand order checked by reading, not run on hardware
//
#include <hip/hip_runtime.h>
#include <hip/hip_bf16.h>

typedef __bf16 bf16_t;
typedef __attribute__((ext_vector_type(16))) __bf16 bf16x16;
typedef __attribute__((ext_vector_type(8)))  __bf16 bf16x8;
typedef __attribute__((ext_vector_type(8)))  float  f32x8;
typedef __attribute__((ext_vector_type(4)))  float  f32x4;

#define TC 32          // timesteps per chunk
#define SEQS 128       // B*R
#define T_TOT 1024

// ---------------- WMMA fragment loaders (CDNA5 16x16x32 bf16 layout) ------
// A (16x32, M x K) and B (32x16 = W[N][K]) use the same per-lane gather:
//   row = lane&15 (M for A, N for B), half = lane>>4
//   elems 0..7  -> K = k0 + half*8 + (0..7)
//   elems 8..15 -> K = k0 + 16 + half*8 + (0..7)
__device__ __forceinline__ bf16x16 frag_ld_bf16(const bf16_t* base,
                                                int row0, int stride, int k0, int lane) {
  const int m = lane & 15, hf = lane >> 4;
  const bf16_t* p = base + (size_t)(row0 + m) * stride + k0 + hf * 8;
  union { bf16x16 v; bf16x8 h[2]; } u;
  u.h[0] = *(const bf16x8*)(p);
  u.h[1] = *(const bf16x8*)(p + 16);
  return u.v;
}

__device__ __forceinline__ bf16x16 frag_ld_f32(const float* base,
                                               int row0, int stride, int k0, int lane) {
  const int m = lane & 15, hf = lane >> 4;
  const float* p = base + (size_t)(row0 + m) * stride + k0 + hf * 8;
  f32x4 a0 = *(const f32x4*)(p);
  f32x4 a1 = *(const f32x4*)(p + 4);
  f32x4 b0 = *(const f32x4*)(p + 16);
  f32x4 b1 = *(const f32x4*)(p + 20);
  bf16x16 v;
#pragma unroll
  for (int i = 0; i < 4; ++i) {
    v[i]      = (bf16_t)a0[i];
    v[4 + i]  = (bf16_t)a1[i];
    v[8 + i]  = (bf16_t)b0[i];
    v[12 + i] = (bf16_t)b1[i];
  }
  return v;
}

__device__ __forceinline__ f32x8 wmma_bf16(bf16x16 a, bf16x16 b, f32x8 c) {
  return __builtin_amdgcn_wmma_f32_16x16x32_bf16(false, a, false, b, (short)0, c,
                                                 false, false);
}

// ---------------- weight conversion (f32 -> bf16 in d_ws) -----------------
// layout in ws (bf16): in_w [4][256][128] | xp_w padded [4][48][128] | out_w [4][128][128]
#define N_INW (4 * 256 * 128)
#define N_XPW (4 * 48 * 128)
#define N_OUTW (4 * 128 * 128)

__global__ void cvt_weights(const float* __restrict__ in_w,
                            const float* __restrict__ xp_w,
                            const float* __restrict__ out_w,
                            bf16_t* __restrict__ ws) {
  int idx = blockIdx.x * blockDim.x + threadIdx.x;
  if (idx < N_INW) { ws[idx] = (bf16_t)in_w[idx]; return; }
  idx -= N_INW;
  if (idx < N_XPW) {
    const int l = idx / (48 * 128);
    const int rem = idx - l * 48 * 128;
    const int r = rem >> 7, k = rem & 127;
    const float v = (r < 40) ? xp_w[(l * 40 + r) * 128 + k] : 0.f;
    ws[N_INW + idx] = (bf16_t)v;
    return;
  }
  idx -= N_XPW;
  if (idx < N_OUTW) ws[N_INW + N_XPW + idx] = (bf16_t)out_w[idx];
}

// ---------------- one fused Mamba layer over a TC-chunk -------------------
__device__ __forceinline__ void run_layer(
    int L, int lane, int wv, int tid,
    float* sh_h, bf16_t* sh_zg, bf16_t* sh_xi, float* sh_dbc, bf16_t* sh_res,
    float* sh_xc, bf16_t* sh_ch,
    const float* ln_g, const float* ln_b,
    const float* conv_w, const float* conv_b,
    const float* dp_w, const float* dp_b,
    const float* A_log, const float* Dp,
    const bf16_t* inwB, const bf16_t* xpwB,
    const bf16_t* outwB, float* st /*8 persistent states*/) {
  // ---- LayerNorm: h -> zg (bf16).  8 threads per row, shfl tree reduce.
  {
    const int row = tid >> 3, q = tid & 7;
    const float* hr = sh_h + row * 128;
    float s = 0.f, s2 = 0.f;
#pragma unroll
    for (int i = 0; i < 16; ++i) { float v = hr[q * 16 + i]; s += v; s2 += v * v; }
    s  += __shfl_xor(s, 1, 32);  s  += __shfl_xor(s, 2, 32);  s  += __shfl_xor(s, 4, 32);
    s2 += __shfl_xor(s2, 1, 32); s2 += __shfl_xor(s2, 2, 32); s2 += __shfl_xor(s2, 4, 32);
    const float mu  = s * (1.f / 128.f);
    const float var = s2 * (1.f / 128.f) - mu * mu;
    const float inv = rsqrtf(var + 1e-5f);
#pragma unroll
    for (int i = 0; i < 16; ++i) {
      const int k = q * 16 + i;
      sh_zg[row * 128 + k] =
          (bf16_t)((hr[k] - mu) * inv * ln_g[L * 128 + k] + ln_b[L * 128 + k]);
    }
  }
  // conv history -> xi rows 0..1 (one element per thread)
  sh_xi[tid] = sh_ch[L * 256 + tid];
  __syncthreads();

  // ---- GEMM1: xr = zg @ in_w^T  (M=32, N=256, K=128) -> xi(bf16) | res(bf16)
#pragma unroll 1
  for (int i = 0; i < 4; ++i) {
    const int tt = wv * 4 + i;
    const int m0 = (tt & 1) * 16;
    const int o0 = (tt >> 1) * 16;
    f32x8 c = {0.f, 0.f, 0.f, 0.f, 0.f, 0.f, 0.f, 0.f};
#pragma unroll
    for (int kk = 0; kk < 128; kk += 32) {
      bf16x16 a = frag_ld_bf16(sh_zg, m0, 128, kk, lane);
      bf16x16 b = frag_ld_bf16(inwB + L * 256 * 128, o0, 128, kk, lane);
      c = wmma_bf16(a, b, c);
    }
    const int col = lane & 15, hf = lane >> 4;
    if (o0 < 128) {
#pragma unroll
      for (int r = 0; r < 8; ++r)
        sh_xi[(m0 + r + 8 * hf + 2) * 128 + o0 + col] = (bf16_t)c[r];
    } else {
#pragma unroll
      for (int r = 0; r < 8; ++r)
        sh_res[(m0 + r + 8 * hf) * 128 + (o0 - 128) + col] = (bf16_t)c[r];
    }
  }
  __syncthreads();

  // ---- depthwise causal conv (k=3) + SiLU -> xc (f32); update history
  {
#pragma unroll
    for (int it = 0; it < 16; ++it) {
      const int idx = it * 256 + tid;
      const int t = idx >> 7, dch = idx & 127;
      const float w0 = conv_w[(L * 128 + dch) * 3 + 0];
      const float w1 = conv_w[(L * 128 + dch) * 3 + 1];
      const float w2 = conv_w[(L * 128 + dch) * 3 + 2];
      float v = (float)sh_xi[t * 128 + dch] * w0 +
                (float)sh_xi[(t + 1) * 128 + dch] * w1 +
                (float)sh_xi[(t + 2) * 128 + dch] * w2 + conv_b[L * 128 + dch];
      sh_xc[idx] = v / (1.f + __expf(-v));   // SiLU
    }
    sh_ch[L * 256 + tid] = sh_xi[32 * 128 + tid];  // last two rows -> history
  }
  __syncthreads();

  // ---- GEMM2: dbc = xc @ xp_w^T (M=32, N=48 padded, K=128), A cvt'd on the fly
  if (wv < 6) {
    const int m0 = (wv & 1) * 16;
    const int n0 = (wv >> 1) * 16;
    f32x8 c = {0.f, 0.f, 0.f, 0.f, 0.f, 0.f, 0.f, 0.f};
#pragma unroll
    for (int kk = 0; kk < 128; kk += 32) {
      bf16x16 a = frag_ld_f32(sh_xc, m0, 128, kk, lane);
      bf16x16 b = frag_ld_bf16(xpwB + L * 48 * 128, n0, 128, kk, lane);
      c = wmma_bf16(a, b, c);
    }
    const int col = lane & 15, hf = lane >> 4;
#pragma unroll
    for (int r = 0; r < 8; ++r)
      sh_dbc[(m0 + r + 8 * hf) * 48 + n0 + col] = c[r];
  }
  __syncthreads();

  // ---- selective scan: thread owns d=tid>>1, n in [ng*8, ng*8+8)
  {
    const int d = tid >> 1;
    const int ng = tid & 1;
    float Ar[8], dw[8];
#pragma unroll
    for (int j = 0; j < 8; ++j)
      Ar[j] = -__expf(A_log[(L * 128 + d) * 16 + ng * 8 + j]);
#pragma unroll
    for (int r = 0; r < 8; ++r) dw[r] = dp_w[(L * 128 + d) * 8 + r];
    const float dpb = dp_b[L * 128 + d];
    const float DpV = Dp[L * 128 + d];
#pragma unroll 1
    for (int t = 0; t < TC; ++t) {
      float acc = dpb;
#pragma unroll
      for (int r = 0; r < 8; ++r) acc += sh_dbc[t * 48 + r] * dw[r];
      const float dt = (acc > 20.f) ? acc : __logf(1.f + __expf(acc));  // softplus
      const float u = sh_xc[t * 128 + d];
      float y = 0.f;
#pragma unroll
      for (int j = 0; j < 8; ++j) {
        const float Bv = sh_dbc[t * 48 + 8 + ng * 8 + j];
        const float Cv = sh_dbc[t * 48 + 24 + ng * 8 + j];
        const float dA = __expf(dt * Ar[j]);
        st[j] = dA * st[j] + (dt * Bv) * u;
        y += st[j] * Cv;
      }
      y += __shfl_xor(y, 1, 32);  // pair (2d, 2d+1) within a wave32
      if (ng == 0) {
        const float yo = y + u * DpV;
        const float rv = (float)sh_res[t * 128 + d];
        const float gate = rv / (1.f + __expf(-rv));  // SiLU(res)
        sh_zg[t * 128 + d] = (bf16_t)(yo * gate);     // reuse zg as ygate
      }
    }
  }
  __syncthreads();

  // ---- GEMM3: h += ygate @ out_w^T (M=32, N=128, K=128)
#pragma unroll 1
  for (int i = 0; i < 2; ++i) {
    const int tt = wv * 2 + i;
    const int m0 = (tt & 1) * 16;
    const int n0 = (tt >> 1) * 16;
    f32x8 c = {0.f, 0.f, 0.f, 0.f, 0.f, 0.f, 0.f, 0.f};
#pragma unroll
    for (int kk = 0; kk < 128; kk += 32) {
      bf16x16 a = frag_ld_bf16(sh_zg, m0, 128, kk, lane);
      bf16x16 b = frag_ld_bf16(outwB + L * 128 * 128, n0, 128, kk, lane);
      c = wmma_bf16(a, b, c);
    }
    const int col = lane & 15, hf = lane >> 4;
#pragma unroll
    for (int r = 0; r < 8; ++r)
      sh_h[(m0 + r + 8 * hf) * 128 + n0 + col] += c[r];  // residual add
  }
  __syncthreads();
}

// ---------------- fully fused model: one workgroup per sequence -----------
__global__ __launch_bounds__(256) void mamba_fused(
    const float* __restrict__ x, const float* __restrict__ inp_w,
    const float* __restrict__ inp_b, const float* __restrict__ outp_w,
    const float* __restrict__ outp_b, const float* __restrict__ ln_g_in,
    const float* __restrict__ ln_b_in, const float* __restrict__ conv_w_in,
    const float* __restrict__ conv_b_in, const float* __restrict__ dp_w_in,
    const float* __restrict__ dp_b_in, const float* __restrict__ A_log_in,
    const float* __restrict__ Dp_in, const bf16_t* __restrict__ inwB_in,
    const bf16_t* __restrict__ xpwB_in, const bf16_t* __restrict__ outwB_in,
    float* __restrict__ out) {
  __shared__ __align__(16) unsigned char smem[60416];
  float*  sh_h   = (float*)(smem + 0);       // [32][128] f32     16384 B
  bf16_t* sh_zg  = (bf16_t*)(smem + 16384);  // [32][128] bf16     8192 B (z / ygate)
  bf16_t* sh_xi  = (bf16_t*)(smem + 24576);  // [34][128] bf16     8704 B
  float*  sh_dbc = (float*)(smem + 24576);   // [32][48]  f32  (aliases xi)
  bf16_t* sh_res = (bf16_t*)(smem + 33280);  // [32][128] bf16     8192 B
  float*  sh_xc  = (float*)(smem + 41472);   // [32][128] f32     16384 B
  bf16_t* sh_ch  = (bf16_t*)(smem + 57856);  // [4][2][128] bf16   2048 B
  float*  sh_x   = (float*)(smem + 59904);   // [32] f32, x slice   128 B

  const int tid = threadIdx.x;
  const int lane = tid & 31;
  const int wv = tid >> 5;
  const int seq = blockIdx.x;

  // per-layer persistent scan states (registers)
  float st0[8], st1[8], st2[8], st3[8];
#pragma unroll
  for (int j = 0; j < 8; ++j) { st0[j] = 0.f; st1[j] = 0.f; st2[j] = 0.f; st3[j] = 0.f; }
  // zero conv history
  for (int i = tid; i < 1024; i += 256) sh_ch[i] = (bf16_t)0.f;
  __syncthreads();

  const float ob = outp_b[0];
  const float iw = inp_w[tid & 127];
  const float ib = inp_b[tid & 127];

  // Opaque zero offset, laundered each chunk iteration.  All weight pointers
  // are derived from the original (address-space-inferable) kernel args plus
  // this offset, so loads stay global_load_* but can no longer be hoisted
  // (and spilled) across the chunk loop by LICM.
  long ofs = 0;

#pragma unroll 1
  for (int c = 0; c < T_TOT / TC; ++c) {
    const int t0 = c * TC;
    asm volatile("" : "+s"(ofs));
    const float*  ln_g   = ln_g_in + ofs;
    const float*  ln_b   = ln_b_in + ofs;
    const float*  conv_w = conv_w_in + ofs;
    const float*  conv_b = conv_b_in + ofs;
    const float*  dp_w   = dp_w_in + ofs;
    const float*  dp_b   = dp_b_in + ofs;
    const float*  A_log  = A_log_in + ofs;
    const float*  Dp     = Dp_in + ofs;
    const bf16_t* inwB   = inwB_in + ofs;
    const bf16_t* xpwB   = xpwB_in + ofs;
    const bf16_t* outwB  = outwB_in + ofs;

    if (t0 + TC < T_TOT)
      __builtin_prefetch(x + (size_t)seq * T_TOT + t0 + TC, 0, 1);

    // stage the 32-float x slice into LDS with the CDNA5 async-DMA path
    if (wv == 0) {
      unsigned ldsa = (unsigned)(unsigned long long)(sh_x + lane);
      unsigned voff = (unsigned)(lane * 4);
      const float* sbase = x + (size_t)seq * T_TOT + t0;
      asm volatile("global_load_async_to_lds_b32 %0, %1, %2"
                   :: "v"(ldsa), "v"(voff), "s"(sbase) : "memory");
      asm volatile("s_wait_asynccnt 0x0" ::: "memory");
    }
    __syncthreads();

    // input projection: h[t][d] = x[seq][t]*inp_w[d] + inp_b[d]
#pragma unroll
    for (int it = 0; it < 16; ++it) {
      const int idx = it * 256 + tid;
      const int t = idx >> 7;
      sh_h[idx] = sh_x[t] * iw + ib;
    }
    __syncthreads();

    run_layer(0, lane, wv, tid, sh_h, sh_zg, sh_xi, sh_dbc, sh_res, sh_xc, sh_ch,
              ln_g, ln_b, conv_w, conv_b, dp_w, dp_b, A_log, Dp, inwB, xpwB, outwB, st0);
    run_layer(1, lane, wv, tid, sh_h, sh_zg, sh_xi, sh_dbc, sh_res, sh_xc, sh_ch,
              ln_g, ln_b, conv_w, conv_b, dp_w, dp_b, A_log, Dp, inwB, xpwB, outwB, st1);
    run_layer(2, lane, wv, tid, sh_h, sh_zg, sh_xi, sh_dbc, sh_res, sh_xc, sh_ch,
              ln_g, ln_b, conv_w, conv_b, dp_w, dp_b, A_log, Dp, inwB, xpwB, outwB, st2);
    run_layer(3, lane, wv, tid, sh_h, sh_zg, sh_xi, sh_dbc, sh_res, sh_xc, sh_ch,
              ln_g, ln_b, conv_w, conv_b, dp_w, dp_b, A_log, Dp, inwB, xpwB, outwB, st3);

    // output projection: out[seq][t] = h[t] . outp_w + outp_b
    {
      const int row = tid >> 3, q = tid & 7;
      float s = 0.f;
#pragma unroll
      for (int i = 0; i < 16; ++i)
        s += sh_h[row * 128 + q * 16 + i] * outp_w[q * 16 + i];
      s += __shfl_xor(s, 1, 32); s += __shfl_xor(s, 2, 32); s += __shfl_xor(s, 4, 32);
      if (q == 0) out[(size_t)seq * T_TOT + t0 + row] = s + ob;
    }
    __syncthreads();  // h reused next chunk
  }
}

extern "C" void kernel_launch(void* const* d_in, const int* in_sizes, int n_in,
                              void* d_out, int out_size, void* d_ws, size_t ws_size,
                              hipStream_t stream) {
  const float* x      = (const float*)d_in[0];
  const float* inp_w  = (const float*)d_in[1];
  const float* inp_b  = (const float*)d_in[2];
  const float* outp_w = (const float*)d_in[3];
  const float* outp_b = (const float*)d_in[4];
  const float* ln_g   = (const float*)d_in[5];
  const float* ln_b   = (const float*)d_in[6];
  const float* in_w   = (const float*)d_in[7];
  const float* conv_w = (const float*)d_in[8];
  const float* conv_b = (const float*)d_in[9];
  const float* xp_w   = (const float*)d_in[10];
  const float* dp_w   = (const float*)d_in[11];
  const float* dp_b   = (const float*)d_in[12];
  const float* A_log  = (const float*)d_in[13];
  const float* Dp     = (const float*)d_in[14];
  const float* out_w  = (const float*)d_in[15];

  bf16_t* wsb = (bf16_t*)d_ws;
  const int n_total = N_INW + N_XPW + N_OUTW;  // 221184 elems, 442368 B in ws
  cvt_weights<<<(n_total + 255) / 256, 256, 0, stream>>>(in_w, xp_w, out_w, wsb);

  const bf16_t* inwB  = wsb;
  const bf16_t* xpwB  = wsb + N_INW;
  const bf16_t* outwB = wsb + N_INW + N_XPW;

  mamba_fused<<<SEQS, 256, 0, stream>>>(x, inp_w, inp_b, outp_w, outp_b, ln_g, ln_b,
                                        conv_w, conv_b, dp_w, dp_b, A_log, Dp,
                                        inwB, xpwB, outwB, (float*)d_out);
}
